// MambaWrap_39084202394358
// MI455X (gfx1250) — matrix operand order repaired and checked by reading, not compiled, not run on hardware
//
#include <hip/hip_runtime.h>

#define D_MODEL   768
#define D_INNER   1536
#define D_STATE   128
#define NHEADS    24
#define HEADDIM   64
#define D_CONV    4
#define CONV_DIM  (D_INNER + 2 * D_STATE)              // 1792
#define D_IN_PROJ (2 * D_INNER + 2 * D_STATE + NHEADS) // 3352
#define BATCH     8
#define SEQLEN    2048
#define TOKENS    (BATCH * SEQLEN)                     // 16384

typedef __attribute__((ext_vector_type(16))) _Float16 v16h;
typedef __attribute__((ext_vector_type(8)))  _Float16 v8h;
typedef __attribute__((ext_vector_type(8)))  float    v8f;

// ---------------------------------------------------------------- utilities
__global__ __launch_bounds__(256) void f32_to_f16_kernel(const float* __restrict__ in,
                                                         _Float16* __restrict__ out, int n) {
    int i = blockIdx.x * 256 + threadIdx.x;
    if (i < n) out[i] = (_Float16)in[i];
}

// ---------------------------------------------------------------- LayerNorm -> f16
__global__ __launch_bounds__(256) void ln_kernel(const float* __restrict__ x,
                                                 const float* __restrict__ ln_w,
                                                 const float* __restrict__ ln_b,
                                                 _Float16* __restrict__ u16) {
    int tok = blockIdx.x;
    int tid = threadIdx.x;
    const float* row = x + (size_t)tok * D_MODEL;
    float v0 = row[tid], v1 = row[tid + 256], v2 = row[tid + 512];
    __shared__ float r1[256], r2[256];
    r1[tid] = v0 + v1 + v2;
    r2[tid] = v0 * v0 + v1 * v1 + v2 * v2;
    __syncthreads();
    for (int off = 128; off > 0; off >>= 1) {
        if (tid < off) { r1[tid] += r1[tid + off]; r2[tid] += r2[tid + off]; }
        __syncthreads();
    }
    float mu   = r1[0] * (1.0f / D_MODEL);
    float var  = r2[0] * (1.0f / D_MODEL) - mu * mu;
    float rstd = rsqrtf(var + 1e-5f);
    _Float16* orow = u16 + (size_t)tok * D_MODEL;
    orow[tid]       = (_Float16)((v0 - mu) * rstd * ln_w[tid]       + ln_b[tid]);
    orow[tid + 256] = (_Float16)((v1 - mu) * rstd * ln_w[tid + 256] + ln_b[tid + 256]);
    orow[tid + 512] = (_Float16)((v2 - mu) * rstd * ln_w[tid + 512] + ln_b[tid + 512]);
}

// ---------------------------------------------------------------- WMMA fragment helpers
__device__ __forceinline__ v16h frag16(const _Float16* __restrict__ p, int hi_off) {
    v8h lo = *(const v8h*)p;
    v8h hi = *(const v8h*)(p + hi_off);
    v16h f;
#pragma unroll
    for (int i = 0; i < 8; ++i) { f[i] = lo[i]; f[i + 8] = hi[i]; }
    return f;
}

__device__ __forceinline__ void wmma_tile(const v16h a[2], const v16h b[4], v8f acc[2][4]) {
#pragma unroll
    for (int j = 0; j < 4; ++j) {
#pragma unroll
        for (int i = 0; i < 2; ++i) {
            acc[i][j] = __builtin_amdgcn_wmma_f32_16x16x32_f16(false, a[i], false, b[j],
                                                               (short)0, acc[i][j], false, false);
        }
    }
}

// ---------------------------------------------------------------- generic f16 GEMM, f32 accum
// C[M][N] = A[M][K] * B[N][K]^T (+ resid).
// Wave computes 32x64 (2x4 WMMA tiles). Block = 8 waves = 64(M) x 256(N) tile.
// Branchless loads (B rows clamped), software pipelined (16 WMMAs / body).
__global__ __launch_bounds__(256) void gemm_f16_kernel(const _Float16* __restrict__ A,
                                                       const _Float16* __restrict__ B,
                                                       float* __restrict__ C,
                                                       const float* __restrict__ resid,
                                                       int N, int K, int lda, int ldb, int ldc) {
    int wave  = threadIdx.x >> 5;
    int lane  = threadIdx.x & 31;
    int waveM = wave & 1;
    int waveN = wave >> 1;
    int mbase = blockIdx.y * 64 + waveM * 32;
    int nbase = blockIdx.x * 256 + waveN * 64;

    const _Float16* aptr[2];
#pragma unroll
    for (int i = 0; i < 2; ++i)
        aptr[i] = A + (size_t)(mbase + i * 16 + (lane & 15)) * lda + (lane & 16);

    const _Float16* bptr[4];
#pragma unroll
    for (int j = 0; j < 4; ++j) {
        int n = nbase + j * 16 + (lane & 15);
        n = (n < N) ? n : (N - 1);
        bptr[j] = B + (size_t)n * ldb + (lane & 16);
    }

    v8f acc[2][4];
#pragma unroll
    for (int i = 0; i < 2; ++i)
#pragma unroll
        for (int j = 0; j < 4; ++j) acc[i][j] = (v8f){};

    v16h a0[2], b0[4], a1[2], b1[4];
#pragma unroll
    for (int i = 0; i < 2; ++i) a0[i] = frag16(aptr[i], 8);
#pragma unroll
    for (int j = 0; j < 4; ++j) b0[j] = frag16(bptr[j], 8);

    for (int k = 0; k < K; k += 64) {
#pragma unroll
        for (int i = 0; i < 2; ++i) a1[i] = frag16(aptr[i] + k + 32, 8);
#pragma unroll
        for (int j = 0; j < 4; ++j) b1[j] = frag16(bptr[j] + k + 32, 8);
        if (k + 128 < K) {
            __builtin_prefetch(aptr[0] + k + 128, 0, 3);   // global_prefetch_b8
            __builtin_prefetch(bptr[0] + k + 128, 0, 3);
        }
        wmma_tile(a0, b0, acc);
        if (k + 64 < K) {
#pragma unroll
            for (int i = 0; i < 2; ++i) a0[i] = frag16(aptr[i] + k + 64, 8);
#pragma unroll
            for (int j = 0; j < 4; ++j) b0[j] = frag16(bptr[j] + k + 64, 8);
        }
        wmma_tile(a1, b1, acc);
    }

    int rowoff = (lane & 16) >> 1;
    int ncol   = lane & 15;
#pragma unroll
    for (int i = 0; i < 2; ++i) {
#pragma unroll
        for (int j = 0; j < 4; ++j) {
            int n = nbase + j * 16 + ncol;
            if (n < N) {
#pragma unroll
                for (int r = 0; r < 8; ++r) {
                    size_t off = (size_t)(mbase + i * 16 + rowoff + r) * ldc + n;
                    float v = acc[i][j][r];
                    if (resid) C[off] = resid[off] + v;
                    else       __builtin_nontemporal_store(v, &C[off]);  // zxbcdt streams past L2
                }
            }
        }
    }
}

// ---------------------------------------------------------------- causal depthwise conv + SiLU
__global__ __launch_bounds__(256) void conv_kernel(const float* __restrict__ zxbcdt,
                                                   const float* __restrict__ conv_w,
                                                   const float* __restrict__ conv_b,
                                                   float* __restrict__ xBCs) {
    int idx = blockIdx.x * 256 + threadIdx.x;
    if (idx >= TOKENS * CONV_DIM) return;
    int c   = idx % CONV_DIM;
    int tok = idx / CONV_DIM;
    int l   = tok & (SEQLEN - 1);
    float s = conv_b[c];
#pragma unroll
    for (int k = 0; k < D_CONV; ++k) {
        int ls = l - (D_CONV - 1) + k;
        if (ls >= 0)
            s += zxbcdt[(size_t)(tok - (D_CONV - 1) + k) * D_IN_PROJ + D_INNER + c] * conv_w[c * D_CONV + k];
    }
    xBCs[(size_t)tok * CONV_DIM + c] = s * (1.0f / (1.0f + expf(-s)));   // SiLU
}

// ---------------------------------------------------------------- dt = softplus(raw+bias), dA = exp(dt*A)
__global__ __launch_bounds__(256) void dt_kernel(const float* __restrict__ zxbcdt,
                                                 const float* __restrict__ dt_bias,
                                                 const float* __restrict__ A_log,
                                                 float* __restrict__ dt, float* __restrict__ dA) {
    int i = blockIdx.x * 256 + threadIdx.x;
    if (i >= TOKENS * NHEADS) return;
    int h   = i % NHEADS;
    int tok = i / NHEADS;
    float raw = zxbcdt[(size_t)tok * D_IN_PROJ + (D_IN_PROJ - NHEADS) + h] + dt_bias[h];
    float dtv = (raw > 20.0f) ? raw : log1pf(expf(raw));
    float A   = -expf(A_log[h]);
    dt[i] = dtv;
    dA[i] = expf(dtv * A);
}

// ---------------------------------------------------------------- sequential SSM scan
// One block per (batch, head). State s[64][128] register-resident:
// tid -> p = tid&63 (row), q = tid>>6 (state-column quarter); thread owns s[p][32q..32q+31].
// q is wave-uniform: each lane stages B/C[32q+lane] from global (coalesced) and
// broadcasts per unrolled step via v_readlane (SGPR broadcast, no LDS crossbar).
// The y-reduction never feeds the recurrence, so it is batched: 8 timesteps of
// partials per LDS buffer, double-buffered -> one barrier per 8 steps.
__device__ __forceinline__ float lane_bcast(float v, int i) {
    return __builtin_bit_cast(float, __builtin_amdgcn_readlane(__builtin_bit_cast(int, v), i));
}

__global__ __launch_bounds__(256) void scan_kernel(const float* __restrict__ xBCs,
                                                   const float* __restrict__ dt,
                                                   const float* __restrict__ dA,
                                                   const float* __restrict__ Dp,
                                                   float* __restrict__ ybuf) {
    int bh   = blockIdx.x;
    int b    = bh / NHEADS;
    int h    = bh % NHEADS;
    int tid  = threadIdx.x;
    int lane = tid & 31;
    int p    = tid & 63;
    int q    = tid >> 6;

    __shared__ float yred[2][8][256];

    float s[32];
#pragma unroll
    for (int i = 0; i < 32; ++i) s[i] = 0.0f;
    float Dh = Dp[h];

    const float* base  = xBCs + (size_t)(b * SEQLEN) * CONV_DIM;
    float*       ybase = ybuf + (size_t)(b * SEQLEN) * D_INNER + h * HEADDIM;
    const float* dtp   = dt + (size_t)(b * SEQLEN) * NHEADS + h;
    const float* dAp   = dA + (size_t)(b * SEQLEN) * NHEADS + h;
    int nX = h * HEADDIM + p;
    int nB = D_INNER + (q << 5) + lane;
    int nC = D_INNER + D_STATE + (q << 5) + lane;

    for (int l0 = 0; l0 < SEQLEN; l0 += 8) {
        int bb = (l0 >> 3) & 1;
        float xvv[8], bvv[8], cvv[8], dtv[8], dAv[8];
#pragma unroll
        for (int u = 0; u < 8; ++u) {                       // batched loads -> one clause
            const float* row = base + (size_t)(l0 + u) * CONV_DIM;
            xvv[u] = row[nX];
            bvv[u] = row[nB];
            cvv[u] = row[nC];
            dtv[u] = dtp[(l0 + u) * NHEADS];                // uniform -> s_load
            dAv[u] = dAp[(l0 + u) * NHEADS];
        }
#pragma unroll
        for (int u = 0; u < 8; ++u) {
            float xdt  = dtv[u] * xvv[u];
            float dA_t = dAv[u];
            float yp = 0.0f;
#pragma unroll
            for (int i = 0; i < 32; ++i) {
                float bi = lane_bcast(bvv[u], i);           // v_readlane -> SGPR
                float ci = lane_bcast(cvv[u], i);
                s[i] = s[i] * dA_t + xdt * bi;
                yp  += s[i] * ci;
            }
            yred[bb][u][tid] = yp;
        }
        __syncthreads();
        if (tid < 64) {
#pragma unroll
            for (int u = 0; u < 8; ++u) {
                const float* yr = yred[bb][u];
                float y = yr[tid] + yr[64 + tid] + yr[128 + tid] + yr[192 + tid];
                // for tid<64, p == tid, so xvv[u] is exactly xs at this output element
                ybase[(size_t)(l0 + u) * D_INNER + tid] = y + Dh * xvv[u];
            }
        }
        // next block writes the other buffer; the following reuse of this buffer is
        // ordered behind the next barrier -> single barrier per 8 steps is safe
    }
}

// ---------------------------------------------------------------- gate (y * silu(z)) + RMSNorm -> f16
__global__ __launch_bounds__(256) void gate_rms_kernel(const float* __restrict__ ybuf,
                                                       const float* __restrict__ zxbcdt,
                                                       const float* __restrict__ norm_w,
                                                       _Float16* __restrict__ yn16) {
    int tok = blockIdx.x;
    int tid = threadIdx.x;
    const float* yrow = ybuf + (size_t)tok * D_INNER;
    const float* zrow = zxbcdt + (size_t)tok * D_IN_PROJ;   // z = cols [0, D_INNER)
    float g[6];
    float sq = 0.0f;
#pragma unroll
    for (int j = 0; j < 6; ++j) {
        int idx = tid + j * 256;
        float z = zrow[idx];
        float v = yrow[idx] * (z * (1.0f / (1.0f + expf(-z))));
        g[j] = v;
        sq += v * v;
    }
    __shared__ float r1[256];
    r1[tid] = sq;
    __syncthreads();
    for (int off = 128; off > 0; off >>= 1) {
        if (tid < off) r1[tid] += r1[tid + off];
        __syncthreads();
    }
    float rr = rsqrtf(r1[0] * (1.0f / D_INNER) + 1e-5f);
    _Float16* orow = yn16 + (size_t)tok * D_INNER;
#pragma unroll
    for (int j = 0; j < 6; ++j) {
        int idx = tid + j * 256;
        orow[idx] = (_Float16)(g[j] * rr * norm_w[idx]);
    }
}

// ---------------------------------------------------------------- launch
extern "C" void kernel_launch(void* const* d_in, const int* in_sizes, int n_in,
                              void* d_out, int out_size, void* d_ws, size_t ws_size,
                              hipStream_t stream) {
    (void)in_sizes; (void)n_in; (void)out_size; (void)ws_size;
    const float* x       = (const float*)d_in[0];
    const float* W_in    = (const float*)d_in[1];
    const float* conv_w  = (const float*)d_in[2];
    const float* conv_b  = (const float*)d_in[3];
    const float* dt_bias = (const float*)d_in[4];
    const float* A_log   = (const float*)d_in[5];
    const float* Dp      = (const float*)d_in[6];
    const float* norm_w  = (const float*)d_in[7];
    const float* W_out   = (const float*)d_in[8];
    const float* ln_w    = (const float*)d_in[9];
    const float* ln_b    = (const float*)d_in[10];
    float* out = (float*)d_out;

    char* ws = (char*)d_ws;
    auto alloc = [&](size_t bytes) -> char* {
        char* p = ws;
        ws += (bytes + 255) & ~(size_t)255;
        return p;
    };
    _Float16* u16    = (_Float16*)alloc((size_t)TOKENS * D_MODEL * 2);
    _Float16* Win16  = (_Float16*)alloc((size_t)D_IN_PROJ * D_MODEL * 2);
    _Float16* Wout16 = (_Float16*)alloc((size_t)D_MODEL * D_INNER * 2);
    float*    zxbcdt = (float*)alloc((size_t)TOKENS * D_IN_PROJ * 4);
    float*    xBCs   = (float*)alloc((size_t)TOKENS * CONV_DIM * 4);
    float*    dtb    = (float*)alloc((size_t)TOKENS * NHEADS * 4);
    float*    dAb    = (float*)alloc((size_t)TOKENS * NHEADS * 4);
    float*    ybuf   = (float*)alloc((size_t)TOKENS * D_INNER * 4);
    _Float16* yn16   = (_Float16*)alloc((size_t)TOKENS * D_INNER * 2);

    // 1. cast weights to f16
    {
        int n1 = D_IN_PROJ * D_MODEL;
        f32_to_f16_kernel<<<(n1 + 255) / 256, 256, 0, stream>>>(W_in, Win16, n1);
        int n2 = D_MODEL * D_INNER;
        f32_to_f16_kernel<<<(n2 + 255) / 256, 256, 0, stream>>>(W_out, Wout16, n2);
    }
    // 2. layernorm + cast
    ln_kernel<<<TOKENS, 256, 0, stream>>>(x, ln_w, ln_b, u16);
    // 3. in-projection GEMM: zxbcdt[16384][3352] = u16 @ Win16^T  (K=768)
    gemm_f16_kernel<<<dim3((D_IN_PROJ + 255) / 256, TOKENS / 64), 256, 0, stream>>>(
        u16, Win16, zxbcdt, nullptr, D_IN_PROJ, D_MODEL, D_MODEL, D_MODEL, D_IN_PROJ);
    // 4. causal conv + silu
    conv_kernel<<<(TOKENS * CONV_DIM + 255) / 256, 256, 0, stream>>>(zxbcdt, conv_w, conv_b, xBCs);
    // 5. dt / dA
    dt_kernel<<<(TOKENS * NHEADS + 255) / 256, 256, 0, stream>>>(zxbcdt, dt_bias, A_log, dtb, dAb);
    // 6. scan (one block per batch*head, state in registers)
    scan_kernel<<<BATCH * NHEADS, 256, 0, stream>>>(xBCs, dtb, dAb, Dp, ybuf);
    // 7. gate + RMSNorm + cast
    gate_rms_kernel<<<TOKENS, 256, 0, stream>>>(ybuf, zxbcdt, norm_w, yn16);
    // 8. out-projection GEMM with fused residual: out = x + yn16 @ Wout16^T  (K=1536)
    gemm_f16_kernel<<<dim3((D_MODEL + 255) / 256, TOKENS / 64), 256, 0, stream>>>(
        yn16, Wout16, out, x, D_MODEL, D_INNER, D_INNER, D_INNER, D_MODEL);
}
